// MultiArticulatorAutoencoder_5437428597424
// MI455X (gfx1250) — matrix operand order repaired and checked by reading, not compiled, not run on hardware
//
#include <hip/hip_runtime.h>
#include <math.h>

// Problem constants (from reference)
#define A_DIM   8
#define IN_DIM  128
#define H_DIM   256
#define H2_DIM  128
#define C_DIM   16
#define L_DIM   128
#define BS_DIM  32768

typedef __bf16 bf16_t;
typedef __attribute__((ext_vector_type(16))) __bf16 v16bf;
typedef __attribute__((ext_vector_type(8)))  __bf16 v8bf;
typedef __attribute__((ext_vector_type(4)))  __bf16 v4bf;
typedef __attribute__((ext_vector_type(8)))  float  v8f;

#define TILE_ELEMS 512   // 32(K) x 16(N) bf16 fragment-major tile
#define SLAB_MAX   8192  // max weight slab (16 n-tiles) in elements

// Per-articulator fragment-major sizes (elements)
#define EW1_PER_A (IN_DIM * H_DIM)        // 32768: kt=4,  nt=16
#define EW2_PER_A (H_DIM * H2_DIM)        // 32768: kt=8,  nt=8
#define EW3_PER_A (H2_DIM * C_DIM)        //  2048: kt=4,  nt=1
#define DW1_PER_A (32 * H2_DIM)           //  4096: kt=1,  nt=8 (K padded 16->32)
#define DW2_PER_A (H2_DIM * H_DIM)        // 32768: kt=4,  nt=16
#define DW3_PER_A (H_DIM * IN_DIM)        // 32768: kt=8,  nt=8

// -------- CDNA5 async global->LDS path (probe-guarded) ---------------------
#if defined(__has_builtin)
#if __has_builtin(__builtin_amdgcn_global_load_async_to_lds_b128)
#define HAVE_ASYNC_LDS 1
#endif
#endif
#ifndef HAVE_ASYNC_LDS
#define HAVE_ASYNC_LDS 0
#endif

#if HAVE_ASYNC_LDS
// Builtin expects pointers to GCC-vector int4 (per clang diagnostic):
//   param0: __attribute__((vector_size(16))) int __device__ *   (global src)
typedef int v4i_gcc __attribute__((vector_size(16)));
typedef __attribute__((address_space(1))) v4i_gcc* async_gptr_t;
typedef __attribute__((address_space(3))) v4i_gcc* async_lptr_t;
#endif

template <int N>
__device__ __forceinline__ void wait_async() {
#if HAVE_ASYNC_LDS
#if __has_builtin(__builtin_amdgcn_s_wait_asynccnt)
  __builtin_amdgcn_s_wait_asynccnt(N);
#else
  asm volatile("s_wait_asynccnt %0" ::"i"(N) : "memory");
#endif
#endif
}

// Copy nchunks*256 chunks of 16B from global to LDS (per-thread 16B chunks).
__device__ __forceinline__ void stage_slab(bf16_t* lds, const bf16_t* gsrc,
                                           int tid, int nchunks) {
#if HAVE_ASYNC_LDS
  async_gptr_t g = (async_gptr_t)gsrc;
  async_lptr_t l = (async_lptr_t)lds;
#pragma unroll
  for (int i = 0; i < 4; ++i)
    if (i < nchunks)
      __builtin_amdgcn_global_load_async_to_lds_b128(g + (i * 256 + tid),
                                                     l + (i * 256 + tid), 0, 0);
#else
  const v8bf* s = (const v8bf*)gsrc;
  v8bf* d = (v8bf*)lds;
#pragma unroll
  for (int i = 0; i < 4; ++i)
    if (i < nchunks) d[i * 256 + tid] = s[i * 256 + tid];
#endif
}

// -------- WMMA helpers -----------------------------------------------------
__device__ __forceinline__ v8f wmma_bf16(v16bf a, v16bf b, v8f c) {
  return __builtin_amdgcn_wmma_f32_16x16x32_bf16(false, a, false, b, (short)0, c,
                                                 false, false);
}

__device__ __forceinline__ v8f zero_v8f() {
  v8f z = {0.f, 0.f, 0.f, 0.f, 0.f, 0.f, 0.f, 0.f};
  return z;
}

// A fragment (16x32 bf16) from row-major LDS: two aligned 16B loads per lane.
__device__ __forceinline__ v16bf lds_load_A(const bf16_t* base, int ld, int lane) {
  const int m  = lane & 15;
  const int kb = (lane & 16) ? 16 : 0;
  const bf16_t* p = base + m * ld + kb;
  v8bf lo = *(const v8bf*)(p);
  v8bf hi = *(const v8bf*)(p + 8);
  return __builtin_shufflevector(lo, hi, 0, 1, 2, 3, 4, 5, 6, 7,
                                 8, 9, 10, 11, 12, 13, 14, 15);
}

// B fragment (32x16 bf16) from fragment-major LDS tile: contiguous per lane.
__device__ __forceinline__ v16bf lds_load_Bfrag(const bf16_t* tilebase, int lane) {
  const bf16_t* p = tilebase + lane * 16;
  v8bf lo = *(const v8bf*)(p);
  v8bf hi = *(const v8bf*)(p + 8);
  return __builtin_shufflevector(lo, hi, 0, 1, 2, 3, 4, 5, 6, 7,
                                 8, 9, 10, 11, 12, 13, 14, 15);
}

__device__ __forceinline__ void store_D_relu_bf16(bf16_t* dst, int ld, int lane,
                                                  v8f acc, const float* bias16) {
  const int n  = lane & 15;
  const int mo = (lane & 16) ? 8 : 0;
  const float bv = bias16[n];
#pragma unroll
  for (int r = 0; r < 8; ++r) {
    float x = acc[r] + bv;
    x = x > 0.f ? x : 0.f;
    dst[(mo + r) * ld + n] = (bf16_t)x;
  }
}

// ---------------------------------------------------------------------------
// Kernel 0: convert f32 weights to bf16 in fragment-major tile order.
// ---------------------------------------------------------------------------
__global__ __launch_bounds__(256) void swizzle_weights_kernel(
    const float* __restrict__ src, bf16_t* __restrict__ dst,
    int Ksrc, int Kdst, int N) {
  const int idx   = blockIdx.x * 256 + threadIdx.x;
  const int per_a = Kdst * N;
  const int a     = idx / per_a;
  const int rem   = idx % per_a;
  const int k     = rem / N;
  const int n     = rem % N;
  const float v = (k < Ksrc) ? src[(size_t)a * Ksrc * N + (size_t)k * N + n] : 0.f;
  const int ntiles = N >> 4;
  const int tile   = (k >> 5) * ntiles + (n >> 4);
  const int lane   = (n & 15) | (k & 16);
  const int e      = k & 15;
  dst[(size_t)a * per_a + tile * TILE_ELEMS + lane * 16 + e] = (bf16_t)v;
}

// ---------------------------------------------------------------------------
// Kernel 1: fused encoder.  One WG = 32 batch rows x 1 articulator.
// Double-buffered async weight-slab pipeline.
// ---------------------------------------------------------------------------
__global__ __launch_bounds__(256) void encoder_kernel(
    const float* __restrict__ x,
    const bf16_t* __restrict__ w1, const float* __restrict__ b1,
    const bf16_t* __restrict__ w2, const float* __restrict__ b2,
    const bf16_t* __restrict__ w3, const float* __restrict__ b3,
    float* __restrict__ z) {
  __shared__ __align__(16) bf16_t xs[32][IN_DIM];     // 8 KB (reused as h2)
  __shared__ __align__(16) bf16_t h1[32][H_DIM];      // 16 KB
  __shared__ __align__(16) bf16_t wbuf[2][SLAB_MAX];  // 32 KB double buffer

  const int tile = blockIdx.x;
  const int a    = blockIdx.y;
  const int row0 = tile * 32;
  const int tid  = threadIdx.x;
  const int lane = tid & 31;
  const int wv   = tid >> 5;
  const int m_t  = wv & 1;

  const bf16_t* w1a = w1 + (size_t)a * EW1_PER_A;
  stage_slab(wbuf[0], w1a, tid, 4);                   // preload GEMM1 slab 0

  // Stage x tile (32x128 f32 -> bf16), vectorized.
#pragma unroll
  for (int i = 0; i < 4; ++i) {
    const int e = (i * 256 + tid) * 4;
    const int r = e >> 7;
    const int c = e & 127;
    const float4 f =
        *(const float4*)&x[(size_t)(row0 + r) * (A_DIM * IN_DIM) + a * IN_DIM + c];
    v4bf t = {(bf16_t)f.x, (bf16_t)f.y, (bf16_t)f.z, (bf16_t)f.w};
    *(v4bf*)&xs[r][c] = t;
  }

  // ---------------- GEMM1: (32x128) @ (128x256) -> h1, relu ----------------
  const int ng = wv >> 1;
  v8f acc1[4];
#pragma unroll
  for (int j = 0; j < 4; ++j) acc1[j] = zero_v8f();

  for (int ks = 0; ks < 4; ++ks) {
    if (ks < 3) {
      stage_slab(wbuf[(ks + 1) & 1], w1a + (ks + 1) * SLAB_MAX, tid, 4);
      wait_async<4>();                 // slab ks resident (in-order completion)
    } else {
      wait_async<0>();
    }
    __syncthreads();
    const bf16_t* wb = wbuf[ks & 1];
    v16bf af = lds_load_A(&xs[m_t * 16][ks * 32], IN_DIM, lane);
#pragma unroll
    for (int j = 0; j < 4; ++j) {
      const int n_t = ng * 4 + j;
      v16bf bfr = lds_load_Bfrag(&wb[n_t * TILE_ELEMS], lane);
      acc1[j] = wmma_bf16(af, bfr, acc1[j]);
    }
    __syncthreads();                   // buffer may be overwritten next iter
  }
#pragma unroll
  for (int j = 0; j < 4; ++j) {
    const int n_t = ng * 4 + j;
    store_D_relu_bf16(&h1[m_t * 16][n_t * 16], H_DIM, lane, acc1[j],
                      b1 + a * H_DIM + n_t * 16);
  }

  // ---------------- GEMM2: (32x256) @ (256x128) -> h2 (=xs), relu ----------
  v8f acc2[2];
#pragma unroll
  for (int j = 0; j < 2; ++j) acc2[j] = zero_v8f();

  const bf16_t* w2a = w2 + (size_t)a * EW2_PER_A;
  stage_slab(wbuf[0], w2a, tid, 2);                   // preload slab 0 (8 KB)
  for (int ks = 0; ks < 8; ++ks) {
    if (ks < 7) {
      stage_slab(wbuf[(ks + 1) & 1], w2a + (ks + 1) * 8 * TILE_ELEMS, tid, 2);
      wait_async<2>();
    } else {
      wait_async<0>();
    }
    __syncthreads();
    const bf16_t* wb = wbuf[ks & 1];
    v16bf af = lds_load_A(&h1[m_t * 16][ks * 32], H_DIM, lane);
#pragma unroll
    for (int j = 0; j < 2; ++j) {
      const int n_t = (wv >> 1) * 2 + j;
      v16bf bfr = lds_load_Bfrag(&wb[n_t * TILE_ELEMS], lane);
      acc2[j] = wmma_bf16(af, bfr, acc2[j]);
    }
    __syncthreads();
  }
  bf16_t* h2 = &xs[0][0];              // reuse xs storage (32x128)
#pragma unroll
  for (int j = 0; j < 2; ++j) {
    const int n_t = (wv >> 1) * 2 + j;
    store_D_relu_bf16(h2 + (m_t * 16) * H2_DIM + n_t * 16, H2_DIM, lane, acc2[j],
                      b2 + a * H2_DIM + n_t * 16);
  }

  // ---------------- GEMM3: (32x128) @ (128x16) -> z (f32) ------------------
  stage_slab(wbuf[0], w3 + (size_t)a * EW3_PER_A, tid, 1);  // all 4 k-tiles
  wait_async<0>();
  __syncthreads();                     // also makes h2 visible

  if (wv < 2) {                        // wave-uniform: EXEC all-ones for WMMA
    v8f acc3 = zero_v8f();
#pragma unroll
    for (int ks = 0; ks < 4; ++ks) {
      v16bf af = lds_load_A(h2 + (wv * 16) * H2_DIM + ks * 32, H2_DIM, lane);
      v16bf bfr = lds_load_Bfrag(&wbuf[0][ks * TILE_ELEMS], lane);
      acc3 = wmma_bf16(af, bfr, acc3);
    }
    const int n  = lane & 15;
    const int mo = (lane & 16) ? 8 : 0;
    const float bv = b3[a * C_DIM + n];
#pragma unroll
    for (int r = 0; r < 8; ++r) {
      const int m = wv * 16 + mo + r;
      z[(size_t)(row0 + m) * (A_DIM * C_DIM) + a * C_DIM + n] = acc3[r] + bv;
    }
  }
}

// ---------------------------------------------------------------------------
// Kernel 2: scatter-max over articulators + tanh -> latent (BS, 128).
// ---------------------------------------------------------------------------
__global__ __launch_bounds__(256) void latent_kernel(
    const float* __restrict__ z, const int* __restrict__ indices,
    float* __restrict__ latent) {
  __shared__ int   idx[A_DIM * C_DIM];
  __shared__ float zs[256];
  const int tid  = threadIdx.x;
  const int gid0 = blockIdx.x * 256;

  if (tid < A_DIM * C_DIM) idx[tid] = indices[tid];
  zs[tid] = z[(size_t)gid0 + tid];
  __syncthreads();

  const int b_local = tid >> 7;
  const int l       = tid & (L_DIM - 1);
  const float* zb = zs + b_local * (A_DIM * C_DIM);
  float m = -INFINITY;
#pragma unroll
  for (int j = 0; j < A_DIM * C_DIM; ++j)
    if (idx[j] == l) m = fmaxf(m, zb[j]);
  latent[(size_t)gid0 + tid] = tanhf(m);
}

// ---------------------------------------------------------------------------
// Kernel 3: fused decoder.  One WG = 32 batch rows x 1 articulator.
// ---------------------------------------------------------------------------
__global__ __launch_bounds__(256) void decoder_kernel(
    const float* __restrict__ latent, const int* __restrict__ indices,
    const bf16_t* __restrict__ w1, const float* __restrict__ b1,
    const bf16_t* __restrict__ w2, const float* __restrict__ b2,
    const bf16_t* __restrict__ w3, const float* __restrict__ b3,
    float* __restrict__ out) {
  __shared__ __align__(16) bf16_t din[32][32];        //  2 KB
  __shared__ __align__(16) bf16_t d1[32][H2_DIM];     //  8 KB
  __shared__ __align__(16) bf16_t d2[32][H_DIM];      // 16 KB
  __shared__ __align__(16) bf16_t wbuf[2][SLAB_MAX];  // 32 KB double buffer
  __shared__ int idx[C_DIM];

  const int tile = blockIdx.x;
  const int a    = blockIdx.y;
  const int row0 = tile * 32;
  const int tid  = threadIdx.x;
  const int lane = tid & 31;
  const int wv   = tid >> 5;
  const int m_t  = wv & 1;

  stage_slab(wbuf[0], w1 + (size_t)a * DW1_PER_A, tid, 2);  // pre-padded W1

  if (tid < C_DIM) idx[tid] = indices[a * C_DIM + tid];
  __syncthreads();

  // Gather dec_in (32x16) from latent, zero-pad cols 16..31.
#pragma unroll
  for (int i = 0; i < 4; ++i) {
    const int e = i * 256 + tid;
    const int r = e >> 5;
    const int c = e & 31;
    float v = 0.f;
    if (c < C_DIM) v = latent[(size_t)(row0 + r) * L_DIM + idx[c]];
    din[r][c] = (bf16_t)v;
  }
  wait_async<0>();
  __syncthreads();

  // ---------------- GEMM A: (32x32) @ (32x128) -> d1, relu -----------------
  {
    v16bf af = lds_load_A(&din[m_t * 16][0], 32, lane);
#pragma unroll
    for (int j = 0; j < 2; ++j) {
      const int n_t = (wv >> 1) * 2 + j;
      v16bf bfr = lds_load_Bfrag(&wbuf[0][n_t * TILE_ELEMS], lane);
      v8f acc = zero_v8f();
      acc = wmma_bf16(af, bfr, acc);
      store_D_relu_bf16(&d1[m_t * 16][n_t * 16], H2_DIM, lane, acc,
                        b1 + a * H2_DIM + n_t * 16);
    }
  }
  __syncthreads();                     // wbuf[0] free for GEMM B preload

  // ---------------- GEMM B: (32x128) @ (128x256) -> d2, relu ---------------
  v8f acc2[4];
#pragma unroll
  for (int j = 0; j < 4; ++j) acc2[j] = zero_v8f();

  const bf16_t* w2a = w2 + (size_t)a * DW2_PER_A;
  stage_slab(wbuf[0], w2a, tid, 4);                   // preload slab 0 (16 KB)
  for (int ks = 0; ks < 4; ++ks) {
    if (ks < 3) {
      stage_slab(wbuf[(ks + 1) & 1], w2a + (ks + 1) * SLAB_MAX, tid, 4);
      wait_async<4>();
    } else {
      wait_async<0>();
    }
    __syncthreads();
    const bf16_t* wb = wbuf[ks & 1];
    v16bf af = lds_load_A(&d1[m_t * 16][ks * 32], H2_DIM, lane);
#pragma unroll
    for (int j = 0; j < 4; ++j) {
      const int n_t = (wv >> 1) * 4 + j;
      v16bf bfr = lds_load_Bfrag(&wb[n_t * TILE_ELEMS], lane);
      acc2[j] = wmma_bf16(af, bfr, acc2[j]);
    }
    __syncthreads();
  }
#pragma unroll
  for (int j = 0; j < 4; ++j) {
    const int n_t = (wv >> 1) * 4 + j;
    store_D_relu_bf16(&d2[m_t * 16][n_t * 16], H_DIM, lane, acc2[j],
                      b2 + a * H_DIM + n_t * 16);
  }

  // ---------------- GEMM C: (32x256) @ (256x128) -> outputs (f32) ----------
  v8f acc3[2];
#pragma unroll
  for (int j = 0; j < 2; ++j) acc3[j] = zero_v8f();

  const bf16_t* w3a = w3 + (size_t)a * DW3_PER_A;
  stage_slab(wbuf[0], w3a, tid, 2);                   // preload slab 0 (8 KB)
  for (int ks = 0; ks < 8; ++ks) {
    if (ks < 7) {
      stage_slab(wbuf[(ks + 1) & 1], w3a + (ks + 1) * 8 * TILE_ELEMS, tid, 2);
      wait_async<2>();
    } else {
      wait_async<0>();
    }
    __syncthreads();
    const bf16_t* wb = wbuf[ks & 1];
    v16bf af = lds_load_A(&d2[m_t * 16][ks * 32], H_DIM, lane);
#pragma unroll
    for (int j = 0; j < 2; ++j) {
      const int n_t = (wv >> 1) * 2 + j;
      v16bf bfr = lds_load_Bfrag(&wb[n_t * TILE_ELEMS], lane);
      acc3[j] = wmma_bf16(af, bfr, acc3[j]);
    }
    __syncthreads();
  }
  const int n  = lane & 15;
  const int mo = (lane & 16) ? 8 : 0;
#pragma unroll
  for (int j = 0; j < 2; ++j) {
    const int n_t = (wv >> 1) * 2 + j;
    const float bv = b3[a * IN_DIM + n_t * 16 + n];
#pragma unroll
    for (int r = 0; r < 8; ++r) {
      const int m = m_t * 16 + mo + r;
      out[(size_t)(row0 + m) * (A_DIM * IN_DIM) + a * IN_DIM + n_t * 16 + n] =
          acc3[j][r] + bv;
    }
  }
}

// ---------------------------------------------------------------------------
extern "C" void kernel_launch(void* const* d_in, const int* in_sizes, int n_in,
                              void* d_out, int out_size, void* d_ws, size_t ws_size,
                              hipStream_t stream) {
  const float* x      = (const float*)d_in[0];
  const float* enc_w1 = (const float*)d_in[1];
  const float* enc_b1 = (const float*)d_in[2];
  const float* enc_w2 = (const float*)d_in[3];
  const float* enc_b2 = (const float*)d_in[4];
  const float* enc_w3 = (const float*)d_in[5];
  const float* enc_b3 = (const float*)d_in[6];
  const float* dec_w1 = (const float*)d_in[7];
  const float* dec_b1 = (const float*)d_in[8];
  const float* dec_w2 = (const float*)d_in[9];
  const float* dec_b2 = (const float*)d_in[10];
  const float* dec_w3 = (const float*)d_in[11];
  const float* dec_b3 = (const float*)d_in[12];
  const int*   indices = (const int*)d_in[13];

  float* out    = (float*)d_out;                          // (BS, A, IN)
  float* latent = out + (size_t)BS_DIM * A_DIM * IN_DIM;  // (BS, L)

  // Workspace: z (f32) then fragment-major bf16 weights.
  float*  z   = (float*)d_ws;                             // BS*A*C f32
  bf16_t* wb  = (bf16_t*)((char*)d_ws + (size_t)BS_DIM * A_DIM * C_DIM * 4);
  bf16_t* ew1 = wb;
  bf16_t* ew2 = ew1 + (size_t)A_DIM * EW1_PER_A;
  bf16_t* ew3 = ew2 + (size_t)A_DIM * EW2_PER_A;
  bf16_t* dw1 = ew3 + (size_t)A_DIM * EW3_PER_A;
  bf16_t* dw2 = dw1 + (size_t)A_DIM * DW1_PER_A;
  bf16_t* dw3 = dw2 + (size_t)A_DIM * DW2_PER_A;

  swizzle_weights_kernel<<<(A_DIM * EW1_PER_A) / 256, 256, 0, stream>>>(
      enc_w1, ew1, IN_DIM, IN_DIM, H_DIM);
  swizzle_weights_kernel<<<(A_DIM * EW2_PER_A) / 256, 256, 0, stream>>>(
      enc_w2, ew2, H_DIM, H_DIM, H2_DIM);
  swizzle_weights_kernel<<<(A_DIM * EW3_PER_A) / 256, 256, 0, stream>>>(
      enc_w3, ew3, H2_DIM, H2_DIM, C_DIM);
  swizzle_weights_kernel<<<(A_DIM * DW1_PER_A) / 256, 256, 0, stream>>>(
      dec_w1, dw1, C_DIM, 32, H2_DIM);
  swizzle_weights_kernel<<<(A_DIM * DW2_PER_A) / 256, 256, 0, stream>>>(
      dec_w2, dw2, H2_DIM, H2_DIM, H_DIM);
  swizzle_weights_kernel<<<(A_DIM * DW3_PER_A) / 256, 256, 0, stream>>>(
      dec_w3, dw3, H_DIM, H_DIM, IN_DIM);

  dim3 grid_ae(BS_DIM / 32, A_DIM);
  encoder_kernel<<<grid_ae, 256, 0, stream>>>(x, ew1, enc_b1, ew2, enc_b2,
                                              ew3, enc_b3, z);
  latent_kernel<<<(BS_DIM * L_DIM) / 256, 256, 0, stream>>>(z, indices, latent);
  decoder_kernel<<<grid_ae, 256, 0, stream>>>(latent, indices, dw1, dec_b1,
                                              dw2, dec_b2, dw3, dec_b3, out);
}
